// GIN_88003879895211
// MI455X (gfx1250) — matrix operand order repaired and checked by reading, not compiled, not run on hardware
//
#include <hip/hip_runtime.h>
#include <math.h>

#define KNBR   12
#define NSTEPS 40
#define HID    32

typedef __attribute__((ext_vector_type(16))) _Float16 v16h;
typedef __attribute__((ext_vector_type(8)))  float    v8f;

union HFrag  { v16h v; _Float16 h[16]; uint4 u[2]; };
union CFrag  { v8f  v; float    f[8];  };

// ---------------------------------------------------------------------------
// one-time f32 -> f16 weight conversion (weights reused by all 1250 tiles)
// ---------------------------------------------------------------------------
__global__ void cvt_f16_kernel(const float* __restrict__ w,
                               _Float16* __restrict__ wh, int count)
{
    int i = blockIdx.x * blockDim.x + threadIdx.x;
    if (i < count) wh[i] = (_Float16)w[i];
}

// ---------------------------------------------------------------------------
// y[16 rows x 32] = GBF(d rows, P pairs) @ W   (feature dim = P*40)
// One wave (32 threads) per 16-node tile. A-fragment built on the fly from
// exp(-(d - f_s)^2/g^2); B = pre-converted f16 weights; f32 WMMA accumulate.
// ---------------------------------------------------------------------------
__global__ __launch_bounds__(32)
void gbf_wmma_kernel(const float* __restrict__ d,       // (n, P)
                     const _Float16* __restrict__ wh,   // (P*40, 32) f16
                     float* __restrict__ y,             // (n, 32)
                     int n, int P, float fstep, float inv_g2)
{
    extern __shared__ float dtab[];                 // 16 * P floats
    const int lane = threadIdx.x;
    const int row0 = blockIdx.x * 16;

    for (int idx = lane; idx < 16 * P; idx += 32) {
        int m = idx / P;
        int p = idx - m * P;
        int r = row0 + m; if (r >= n) r = n - 1;
        dtab[idx] = d[(size_t)r * P + p];
    }
    __syncthreads();  // single-wave block: S_NOP, keeps LDS ordering explicit

    const int m  = lane & 15;
    const int hi = lane >> 4;
    const int nchunks = (P * NSTEPS) / 32;

    v8f acc0 = {}; v8f acc1 = {};
    for (int ch = 0; ch < nchunks; ++ch) {
        const int k0 = ch * 32;

        // ---- A fragment: 16x32 f16, documented CDNA5 layout ----
        HFrag a;
        #pragma unroll
        for (int e = 0; e < 16; ++e) {
            int v  = e >> 1, h = e & 1;
            int kl = ((v & 4) << 2) + ((v & 3) << 1) + hi * 8 + h;
            int kg = k0 + kl;
            int p  = kg / NSTEPS;
            int s  = kg - p * NSTEPS;
            float t = dtab[m * P + p] - (float)s * fstep;
            a.h[e] = (_Float16)__expf(-t * t * inv_g2);
        }

        // ---- B fragments: lane&15 = N column, 16 halves = K rows (same k order as A) --
        const _Float16* wp = wh + (size_t)k0 * HID + (lane & 15);
        HFrag b0, b1;
        #pragma unroll
        for (int e = 0; e < 16; ++e) {
            int v  = e >> 1, h = e & 1;
            int kl = ((v & 4) << 2) + ((v & 3) << 1) + hi * 8 + h;
            b0.h[e] = wp[(size_t)kl * HID];        // channels 0..15
            b1.h[e] = wp[(size_t)kl * HID + 16];   // channels 16..31
        }

        acc0 = __builtin_amdgcn_wmma_f32_16x16x32_f16(false, a.v, false, b0.v,
                                                      (short)0, acc0, false, false);
        acc1 = __builtin_amdgcn_wmma_f32_16x16x32_f16(false, a.v, false, b1.v,
                                                      (short)0, acc1, false, false);
    }

    // ---- store D (16x16 f32 layout: VGPR r -> M = r + hi*8, N = lane&15) ----
    const int ncol = lane & 15;
    const int mb   = hi * 8;
    CFrag c0, c1; c0.v = acc0; c1.v = acc1;
    if (row0 + 16 <= n) {               // uniform: full tile, branchless stores
        #pragma unroll
        for (int r = 0; r < 8; ++r) {
            size_t row = (size_t)(row0 + mb + r);
            y[row * HID + ncol]      = c0.f[r];
            y[row * HID + 16 + ncol] = c1.f[r];
        }
    } else {
        #pragma unroll
        for (int r = 0; r < 8; ++r) {
            int row = row0 + mb + r;
            if (row < n) {
                y[(size_t)row * HID + ncol]      = c0.f[r];
                y[(size_t)row * HID + 16 + ncol] = c1.f[r];
            }
        }
    }
}

// ---------------------------------------------------------------------------
// agg[nbr[i,j]] += y[i]   (agg must be pre-initialized to y)
// ---------------------------------------------------------------------------
__global__ void scatter_add_kernel(const int* __restrict__ nbr,
                                   const float* __restrict__ y,
                                   float* __restrict__ agg, int n)
{
    int i = blockIdx.x * blockDim.x + threadIdx.x;
    if (i >= n * HID) return;
    int node = i >> 5;
    int c    = i & 31;
    float v  = y[i];
    #pragma unroll
    for (int j = 0; j < KNBR; ++j) {
        int mdst = nbr[node * KNBR + j];
        atomicAdd(agg + (size_t)mdst * HID + c, v);
    }
}

// ---------------------------------------------------------------------------
// t = relu(agg + b1); u = t @ w2 + b2; out = bn(relu(u))
// ---------------------------------------------------------------------------
__global__ void gin_post_kernel(const float* __restrict__ agg,
                                const float* __restrict__ b1,
                                const float* __restrict__ w2,
                                const float* __restrict__ b2,
                                const float* __restrict__ bn,
                                float* __restrict__ out, int n)
{
    int nidx = blockIdx.x * blockDim.x + threadIdx.x;
    if (nidx >= n) return;
    float t[HID];
    #pragma unroll
    for (int c = 0; c < HID; ++c) {
        float v = agg[(size_t)nidx * HID + c] + b1[c];
        t[c] = v > 0.f ? v : 0.f;
    }
    for (int co = 0; co < HID; ++co) {
        float s = b2[co];
        #pragma unroll
        for (int ci = 0; ci < HID; ++ci) s += t[ci] * w2[ci * HID + co];
        s = s > 0.f ? s : 0.f;
        float g = bn[co], b = bn[32 + co], mu = bn[64 + co], var = bn[96 + co];
        out[(size_t)nidx * HID + co] = (s - mu) * (g * rsqrtf(var + 1e-5f)) + b;
    }
}

// ---------------------------------------------------------------------------
// y = x @ w  (32x32, no bias — bias applied post-aggregation)
// ---------------------------------------------------------------------------
__global__ void mm32_kernel(const float* __restrict__ x,
                            const float* __restrict__ w,
                            float* __restrict__ y, int n)
{
    int nidx = blockIdx.x * blockDim.x + threadIdx.x;
    if (nidx >= n) return;
    float xr[HID];
    #pragma unroll
    for (int c = 0; c < HID; ++c) xr[c] = x[(size_t)nidx * HID + c];
    for (int co = 0; co < HID; ++co) {
        float s = 0.f;
        #pragma unroll
        for (int ci = 0; ci < HID; ++ci) s += xr[ci] * w[ci * HID + co];
        y[(size_t)nidx * HID + co] = s;
    }
}

// ---------------------------------------------------------------------------
// crys[n,16] = [ (hb@f1bw+f1bb)@f2bw+f2bb , (ha@f1aw+f1ab)@f2aw+f2ab ]
// ---------------------------------------------------------------------------
__global__ void head_kernel(const float* __restrict__ hb, const float* __restrict__ ha,
                            const float* __restrict__ f1bw, const float* __restrict__ f1bb,
                            const float* __restrict__ f2bw, const float* __restrict__ f2bb,
                            const float* __restrict__ f1aw, const float* __restrict__ f1ab,
                            const float* __restrict__ f2aw, const float* __restrict__ f2ab,
                            float* __restrict__ crys, int n)
{
    int nidx = blockIdx.x * blockDim.x + threadIdx.x;
    if (nidx >= n) return;
    float x[HID], v[16];

    #pragma unroll
    for (int i = 0; i < HID; ++i) x[i] = hb[(size_t)nidx * HID + i];
    for (int o = 0; o < 16; ++o) {
        float s = f1bb[o];
        #pragma unroll
        for (int i = 0; i < HID; ++i) s += x[i] * f1bw[i * 16 + o];
        v[o] = s;
    }
    for (int o = 0; o < 8; ++o) {
        float s = f2bb[o];
        #pragma unroll
        for (int i = 0; i < 16; ++i) s += v[i] * f2bw[i * 8 + o];
        crys[(size_t)nidx * 16 + o] = s;
    }

    #pragma unroll
    for (int i = 0; i < HID; ++i) x[i] = ha[(size_t)nidx * HID + i];
    for (int o = 0; o < 16; ++o) {
        float s = f1ab[o];
        #pragma unroll
        for (int i = 0; i < HID; ++i) s += x[i] * f1aw[i * 16 + o];
        v[o] = s;
    }
    for (int o = 0; o < 8; ++o) {
        float s = f2ab[o];
        #pragma unroll
        for (int i = 0; i < 16; ++i) s += v[i] * f2aw[i * 8 + o];
        crys[(size_t)nidx * 16 + 8 + o] = s;
    }
}

// ---------------------------------------------------------------------------
// segment mean over [start,end) rows of crys (16 feats), then @ fc_w + fc_b
// one block per segment, 128 threads (8 row-groups x 16 features)
// ---------------------------------------------------------------------------
__global__ void pool_head_kernel(const float* __restrict__ crys,
                                 const int* __restrict__ cidx,
                                 const float* __restrict__ fcw,
                                 const float* __restrict__ fcb,
                                 float* __restrict__ out)
{
    __shared__ float red[128];
    int seg   = blockIdx.x;
    int start = cidx[seg * 2];
    int end   = cidx[seg * 2 + 1];
    int f  = threadIdx.x & 15;
    int rg = threadIdx.x >> 4;   // 0..7
    float s = 0.f;
    for (int r = start + rg; r < end; r += 8) s += crys[(size_t)r * 16 + f];
    red[threadIdx.x] = s;
    __syncthreads();
    if (rg == 0) {
        float tot = 0.f;
        #pragma unroll
        for (int g = 0; g < 8; ++g) tot += red[g * 16 + f];
        red[f] = tot / (float)(end - start);
    }
    __syncthreads();
    if (threadIdx.x < 2) {
        float o = fcb[threadIdx.x];
        #pragma unroll
        for (int i = 0; i < 16; ++i) o += red[i] * fcw[i * 2 + threadIdx.x];
        out[seg * 2 + threadIdx.x] = o;
    }
}

// ---------------------------------------------------------------------------
extern "C" void kernel_launch(void* const* d_in, const int* in_sizes, int n_in,
                              void* d_out, int out_size, void* d_ws, size_t ws_size,
                              hipStream_t stream)
{
    const float* bond    = (const float*)d_in[0];
    const float* angle   = (const float*)d_in[1];
    const float* w_b1_1  = (const float*)d_in[2];
    const float* b_b1_1  = (const float*)d_in[3];
    const float* w_b1_2  = (const float*)d_in[4];
    const float* b_b1_2  = (const float*)d_in[5];
    const float* w_a1_1  = (const float*)d_in[6];
    const float* b_a1_1  = (const float*)d_in[7];
    const float* w_a1_2  = (const float*)d_in[8];
    const float* b_a1_2  = (const float*)d_in[9];
    const float* bn1b    = (const float*)d_in[10];
    const float* bn1a    = (const float*)d_in[11];
    const float* w_b2_1  = (const float*)d_in[12];
    const float* b_b2_1  = (const float*)d_in[13];
    const float* w_b2_2  = (const float*)d_in[14];
    const float* b_b2_2  = (const float*)d_in[15];
    const float* w_a2_1  = (const float*)d_in[16];
    const float* b_a2_1  = (const float*)d_in[17];
    const float* w_a2_2  = (const float*)d_in[18];
    const float* b_a2_2  = (const float*)d_in[19];
    const float* bn2b    = (const float*)d_in[20];
    const float* bn2a    = (const float*)d_in[21];
    const float* fc1b_w  = (const float*)d_in[22];
    const float* fc1b_b  = (const float*)d_in[23];
    const float* fc1a_w  = (const float*)d_in[24];
    const float* fc1a_b  = (const float*)d_in[25];
    const float* fc2b_w  = (const float*)d_in[26];
    const float* fc2b_b  = (const float*)d_in[27];
    const float* fc2a_w  = (const float*)d_in[28];
    const float* fc2a_b  = (const float*)d_in[29];
    const float* fc_w    = (const float*)d_in[30];
    const float* fc_b    = (const float*)d_in[31];
    const int*   nbr     = (const int*)d_in[32];
    const int*   cidx    = (const int*)d_in[33];
    float*       out     = (float*)d_out;

    const int n  = in_sizes[0] / KNBR;       // 20000
    const int ns = in_sizes[33] / 2;         // 200 segments
    const size_t SZ = (size_t)n * HID;

    float* buf0 = (float*)d_ws;
    float* buf1 = buf0 + SZ;
    float* buf2 = buf1 + SZ;
    float* buf3 = buf2 + SZ;
    _Float16* whb = (_Float16*)(buf3 + SZ);                 // 480*32 f16
    _Float16* wha = whb + (size_t)KNBR * NSTEPS * HID;      // 5760*32 f16

    const int ntiles  = (n + 15) / 16;
    const int nthr    = 256;
    const int blks_nc = (n * HID + nthr - 1) / nthr;   // per (node, channel)
    const int blks_n  = (n + nthr - 1) / nthr;         // per node

    const int cb = KNBR * NSTEPS * HID;                // 15360
    const int ca = KNBR * KNBR * NSTEPS * HID;         // 184320

    const float PI = 3.14159265358979323846f;
    const float b_fstep = 8.0f / 39.0f,  b_invg2 = (40.0f / 8.0f) * (40.0f / 8.0f);
    const float a_fstep = PI   / 39.0f,  a_invg2 = (40.0f / PI)   * (40.0f / PI);

    // ---- pre-convert layer-1 weights to f16 (once; reused by all tiles) ----
    cvt_f16_kernel<<<(cb + nthr - 1) / nthr, nthr, 0, stream>>>(w_b1_1, whb, cb);
    cvt_f16_kernel<<<(ca + nthr - 1) / nthr, nthr, 0, stream>>>(w_a1_1, wha, ca);

    // ---- Layer 1 projections: y = GBF(x) @ W1 (WMMA) ----
    gbf_wmma_kernel<<<ntiles, 32, 16 * KNBR * sizeof(float), stream>>>(
        bond, whb, buf0, n, KNBR, b_fstep, b_invg2);
    gbf_wmma_kernel<<<ntiles, 32, 16 * KNBR * KNBR * sizeof(float), stream>>>(
        angle, wha, buf1, n, KNBR * KNBR, a_fstep, a_invg2);

    // ---- Layer 1 aggregation: agg = y + scatter(y) ----
    hipMemcpyAsync(buf2, buf0, 2 * SZ * sizeof(float), hipMemcpyDeviceToDevice, stream);
    scatter_add_kernel<<<blks_nc, nthr, 0, stream>>>(nbr, buf0, buf2, n);
    scatter_add_kernel<<<blks_nc, nthr, 0, stream>>>(nbr, buf1, buf3, n);

    // ---- Layer 1 post: relu/W2/relu/BN ----
    gin_post_kernel<<<blks_n, nthr, 0, stream>>>(buf2, b_b1_1, w_b1_2, b_b1_2, bn1b, buf0, n);
    gin_post_kernel<<<blks_n, nthr, 0, stream>>>(buf3, b_a1_1, w_a1_2, b_a1_2, bn1a, buf1, n);

    // ---- Layer 2 projection ----
    mm32_kernel<<<blks_n, nthr, 0, stream>>>(buf0, w_b2_1, buf2, n);
    mm32_kernel<<<blks_n, nthr, 0, stream>>>(buf1, w_a2_1, buf3, n);

    // ---- Layer 2 aggregation ----
    hipMemcpyAsync(buf0, buf2, 2 * SZ * sizeof(float), hipMemcpyDeviceToDevice, stream);
    scatter_add_kernel<<<blks_nc, nthr, 0, stream>>>(nbr, buf2, buf0, n);
    scatter_add_kernel<<<blks_nc, nthr, 0, stream>>>(nbr, buf3, buf1, n);

    // ---- Layer 2 post ----
    gin_post_kernel<<<blks_n, nthr, 0, stream>>>(buf0, b_b2_1, w_b2_2, b_b2_2, bn2b, buf2, n);
    gin_post_kernel<<<blks_n, nthr, 0, stream>>>(buf1, b_a2_1, w_a2_2, b_a2_2, bn2a, buf3, n);

    // ---- FC head -> crys features (n,16) in buf0 ----
    head_kernel<<<blks_n, nthr, 0, stream>>>(buf2, buf3,
        fc1b_w, fc1b_b, fc2b_w, fc2b_b,
        fc1a_w, fc1a_b, fc2a_w, fc2a_b, buf0, n);

    // ---- segment mean + final 16->2 ----
    pool_head_kernel<<<ns, 128, 0, stream>>>(buf0, cidx, fc_w, fc_b, out);
}
